// GCN_51187420234464
// MI455X (gfx1250) — compile-verified, matrix operand order checked by reading, not run on hardware
//
#include <hip/hip_runtime.h>
#include <hip/hip_bf16.h>
#include <stdint.h>

typedef __attribute__((ext_vector_type(16))) __bf16 v16bf;
typedef __attribute__((ext_vector_type(8)))  float  v8f;

#define NN 50000
#define NE 800000
#define DH 128

// ---------- helpers ----------
__device__ __forceinline__ unsigned short f32_to_bf16_rne(float f) {
  union { float f; unsigned u; } v; v.f = f;
  unsigned u = v.u;
  unsigned r = u + 0x7fffu + ((u >> 16) & 1u);
  if ((u & 0x7fffffffu) > 0x7f800000u) r = u | 0x00400000u; // NaN stays NaN
  return (unsigned short)(r >> 16);
}

// ---------- conversions ----------
__global__ void k_f32_to_bf16(const float* __restrict__ in,
                              unsigned short* __restrict__ out, int n) {
  int i = blockIdx.x * blockDim.x + threadIdx.x;
  if (i < n) out[i] = f32_to_bf16_rne(in[i]);
}

// W: [128, nout] row-major f32  ->  Wt: [npad, 128] bf16, Wt[n][k] = W[k][n], pad rows zero
__global__ void k_w_transpose_bf16(const float* __restrict__ W,
                                   unsigned short* __restrict__ Wt,
                                   int nout, int npad) {
  int i = blockIdx.x * blockDim.x + threadIdx.x;
  if (i >= npad * DH) return;
  int n = i >> 7, k = i & (DH - 1);
  float v = (n < nout) ? W[k * nout + n] : 0.0f;
  Wt[n * DH + k] = f32_to_bf16_rne(v);
}

// ---------- degree / normalization ----------
__global__ void k_deg_init(float* __restrict__ deg) {
  int i = blockIdx.x * blockDim.x + threadIdx.x;
  if (i < NN) deg[i] = 1.0f;                 // self-loop
}
__global__ void k_deg_edges(const long long* __restrict__ dst,
                            float* __restrict__ deg) {
  int e = blockIdx.x * blockDim.x + threadIdx.x;
  if (e < NE) atomicAdd(&deg[(int)dst[e]], 1.0f);
}
__global__ void k_dinv(const float* __restrict__ deg, float* __restrict__ dinv) {
  int i = blockIdx.x * blockDim.x + threadIdx.x;
  if (i < NN) dinv[i] = rsqrtf(deg[i]);      // deg >= 1 always (self-loop)
}

// ---------- WMMA GEMM: C[row,col] = sum_k A[row,k]*Bt[col,k] ----------
// A: [nrows,128] bf16 row-major. Bt: [npad,128] bf16 (pre-transposed weights).
// One wave computes one 16x16 tile; waves in a block cover column tiles.
__global__ void k_gemm_wmma_bf16(const unsigned short* __restrict__ A,
                                 const unsigned short* __restrict__ Bt,
                                 float* __restrict__ C, int nout, int ldc) {
  const int lane  = threadIdx.x & 31;
  const int wave  = threadIdx.x >> 5;
  const int waves = blockDim.x >> 5;
  const int row0  = blockIdx.x << 4;
  const int n0    = (blockIdx.y * waves + wave) << 4;
  const int half  = lane >> 4;     // K-half selector per ISA 16-bit A layout
  const int m     = lane & 15;

  const unsigned short* ap = A  + (size_t)(row0 + m) * DH + half * 8;
  const unsigned short* bp = Bt + (size_t)(n0  + m) * DH + half * 8;

  v8f acc = {};
#pragma unroll
  for (int kc = 0; kc < DH; kc += 32) {
    v16bf a, b;
    ((uint4*)&a)[0] = *(const uint4*)(ap + kc);        // K kc..kc+7   (this half)
    ((uint4*)&a)[1] = *(const uint4*)(ap + kc + 16);   // K kc+16..23  (this half)
    ((uint4*)&b)[0] = *(const uint4*)(bp + kc);
    ((uint4*)&b)[1] = *(const uint4*)(bp + kc + 16);
    acc = __builtin_amdgcn_wmma_f32_16x16x32_bf16(false, a, false, b,
                                                  (short)0, acc, false, false);
  }
  const int col = n0 + m;
  if (col < nout) {
#pragma unroll
    for (int r = 0; r < 8; ++r) {
      int row = row0 + r + half * 8;   // C layout: vgpr r -> M=r (lanes<16) / M=r+8
      C[(size_t)row * ldc + col] = acc[r];
    }
  }
}

// ---------- aggregation ----------
// out[i,:] = dinv[i]^2 * xw[i,:]   (self-loop term, also zero-initializes out)
__global__ void k_self_init(const float* __restrict__ xw,
                            const float* __restrict__ dinv,
                            float* __restrict__ out, int d, int total) {
  int i = blockIdx.x * blockDim.x + threadIdx.x;
  if (i >= total) return;
  float s = dinv[i / d];
  out[i] = s * s * xw[i];
}

// one wave per edge, float4 per lane (d = 128)
__global__ void k_edge_agg_128(const float* __restrict__ xw,
                               const float* __restrict__ dinv,
                               const long long* __restrict__ src,
                               const long long* __restrict__ dst,
                               float* __restrict__ out) {
  unsigned gid = blockIdx.x * blockDim.x + threadIdx.x;
  unsigned e = gid >> 5;
  if (e >= NE) return;
  int lane = gid & 31;
  int s = (int)src[e], t = (int)dst[e];
  float nf = dinv[s] * dinv[t];
  const float4 hv = *(const float4*)(xw + (size_t)s * DH + lane * 4);
  float* op = out + (size_t)t * DH + lane * 4;
  atomicAdd(op + 0, nf * hv.x);
  atomicAdd(op + 1, nf * hv.y);
  atomicAdd(op + 2, nf * hv.z);
  atomicAdd(op + 3, nf * hv.w);
}

// one wave per edge, one lane per feature (d <= 32)
__global__ void k_edge_agg_small(const float* __restrict__ xw,
                                 const float* __restrict__ dinv,
                                 const long long* __restrict__ src,
                                 const long long* __restrict__ dst,
                                 float* __restrict__ out, int d) {
  unsigned gid = blockIdx.x * blockDim.x + threadIdx.x;
  unsigned e = gid >> 5;
  int lane = gid & 31;
  if (e >= NE || lane >= d) return;
  int s = (int)src[e], t = (int)dst[e];
  float nf = dinv[s] * dinv[t];
  atomicAdd(&out[(size_t)t * d + lane], nf * xw[(size_t)s * d + lane]);
}

// h += bias (+ ReLU); also emit bf16 copy for the next layer's WMMA GEMM
__global__ void k_bias_act_bf16(float* __restrict__ h, const float* __restrict__ b,
                                unsigned short* __restrict__ hb, int relu) {
  int i = blockIdx.x * blockDim.x + threadIdx.x;
  if (i >= NN * DH) return;
  float v = h[i] + b[i & (DH - 1)];
  if (relu) v = fmaxf(v, 0.0f);
  h[i]  = v;
  hb[i] = f32_to_bf16_rne(v);
}

// final: o[row,:] = log_softmax(o[row,:] + b)
__global__ void k_bias_logsoftmax(float* __restrict__ o, const float* __restrict__ b, int d) {
  int row = blockIdx.x * blockDim.x + threadIdx.x;
  if (row >= NN) return;
  float v[16];
  float mx = -INFINITY;
  for (int c = 0; c < d; ++c) { v[c] = o[(size_t)row * d + c] + b[c]; mx = fmaxf(mx, v[c]); }
  float s = 0.0f;
  for (int c = 0; c < d; ++c) s += __expf(v[c] - mx);
  float ls = __logf(s) + mx;
  for (int c = 0; c < d; ++c) o[(size_t)row * d + c] = v[c] - ls;
}

// ---------- launch ----------
extern "C" void kernel_launch(void* const* d_in, const int* in_sizes, int n_in,
                              void* d_out, int out_size, void* d_ws, size_t ws_size,
                              hipStream_t stream) {
  (void)in_sizes; (void)n_in; (void)out_size; (void)ws_size;
  const float*      x  = (const float*)d_in[0];
  const long long*  ei = (const long long*)d_in[1];   // [2, NE] int64
  const float*      W1 = (const float*)d_in[2];
  const float*      b1 = (const float*)d_in[3];
  const float*      W2 = (const float*)d_in[4];
  const float*      b2 = (const float*)d_in[5];
  const float*      W3 = (const float*)d_in[6];
  const float*      b3 = (const float*)d_in[7];
  float* out = (float*)d_out;                         // [NN, 10]
  const long long* src = ei;
  const long long* dst = ei + NE;

  char* ws = (char*)d_ws;
  size_t off = 0;
  auto alloc = [&](size_t bytes) -> void* {
    off = (off + 255) & ~(size_t)255;
    void* p = ws + off; off += bytes; return p;
  };
  float*          deg  = (float*)alloc((size_t)NN * 4);
  float*          dinv = (float*)alloc((size_t)NN * 4);
  unsigned short* xb   = (unsigned short*)alloc((size_t)NN * DH * 2);
  unsigned short* hb   = (unsigned short*)alloc((size_t)NN * DH * 2);
  unsigned short* Wt1  = (unsigned short*)alloc((size_t)DH * DH * 2);
  unsigned short* Wt2  = (unsigned short*)alloc((size_t)DH * DH * 2);
  unsigned short* Wt3  = (unsigned short*)alloc((size_t)16 * DH * 2);
  float*          xw   = (float*)alloc((size_t)NN * DH * 4);
  float*          h    = (float*)alloc((size_t)NN * DH * 4);
  float*          xw3  = (float*)alloc((size_t)NN * 16 * 4);

  const int T = 256;
  const int gN   = (NN + T - 1) / T;
  const int gE   = (NE + T - 1) / T;
  const int gND  = (NN * DH + T - 1) / T;
  const int gEW  = (NE * 32 + T - 1) / T;     // wave-per-edge grids
  const int gN10 = (NN * 10 + T - 1) / T;

  // normalization factors
  k_deg_init <<<gN, T, 0, stream>>>(deg);
  k_deg_edges<<<gE, T, 0, stream>>>(dst, deg);
  k_dinv     <<<gN, T, 0, stream>>>(deg, dinv);

  // bf16 operands for WMMA
  k_f32_to_bf16     <<<gND, T, 0, stream>>>(x, xb, NN * DH);
  k_w_transpose_bf16<<<(DH * DH + T - 1) / T, T, 0, stream>>>(W1, Wt1, DH, DH);
  k_w_transpose_bf16<<<(DH * DH + T - 1) / T, T, 0, stream>>>(W2, Wt2, DH, DH);
  k_w_transpose_bf16<<<(16 * DH + T - 1) / T, T, 0, stream>>>(W3, Wt3, 10, 16);

  // ----- layer 1: GCN(128->128) + ReLU -----
  k_gemm_wmma_bf16<<<dim3(NN / 16, 1), 256, 0, stream>>>(xb, Wt1, xw, DH, DH);
  k_self_init     <<<gND, T, 0, stream>>>(xw, dinv, h, DH, NN * DH);
  k_edge_agg_128  <<<gEW, T, 0, stream>>>(xw, dinv, src, dst, h);
  k_bias_act_bf16 <<<gND, T, 0, stream>>>(h, b1, hb, 1);

  // ----- layer 2: GCN(128->128) -----
  k_gemm_wmma_bf16<<<dim3(NN / 16, 1), 256, 0, stream>>>(hb, Wt2, xw, DH, DH);
  k_self_init     <<<gND, T, 0, stream>>>(xw, dinv, h, DH, NN * DH);
  k_edge_agg_128  <<<gEW, T, 0, stream>>>(xw, dinv, src, dst, h);
  k_bias_act_bf16 <<<gND, T, 0, stream>>>(h, b2, hb, 0);

  // ----- layer 3: GCN(128->10) + log_softmax -----
  k_gemm_wmma_bf16<<<dim3(NN / 16, 1), 32, 0, stream>>>(hb, Wt3, xw3, 10, 10);
  k_self_init      <<<gN10, T, 0, stream>>>(xw3, dinv, out, 10, NN * 10);
  k_edge_agg_small <<<gEW, T, 0, stream>>>(xw3, dinv, src, dst, out, 10);
  k_bias_logsoftmax<<<gN, T, 0, stream>>>(out, b3, 10);
}
